// NTSClassifyModel_58763742544674
// MI455X (gfx1250) — matrix operand order repaired and checked by reading, not compiled, not run on hardware
//
#include <hip/hip_runtime.h>
#include <hip/hip_bf16.h>

// ---------------------------------------------------------------------------
// NTS proposal net + NMS + crop-resize for MI455X (gfx1250, wave32, WMMA).
// conv1 (2048->128, 3x3, 14x14, B=16) = implicit-GEMM bf16 WMMA,
// K reordered tap-major so each 32-wide K-block is a single conv tap and
// channel-contiguous -> vector loads, no per-element decode.
// ---------------------------------------------------------------------------

typedef __attribute__((ext_vector_type(16))) __bf16 v16bf;
typedef __attribute__((ext_vector_type(8)))  __bf16 v8bf;
typedef __attribute__((ext_vector_type(8)))  float  v8f;
typedef __attribute__((ext_vector_type(8)))  unsigned short u16x8;

#define BATCH   16
#define HW1     196     // 14*14
#define CIN1    2048
#define NCH     128
#define KDIM    18432   // 2048*9
#define MROWS   3136    // 16*196
#define KSTEPS  576     // KDIM/32
#define NANCH   1614
#define TOPN    6
#define CROP    224
#define IMG     448
#define PADV    224
#define HP      896

#define OUT_IMG_ELEMS  (BATCH*TOPN*3*CROP*CROP)   // 14450688
#define OUT_IDX_OFF    OUT_IMG_ELEMS
#define OUT_PROB_OFF   (OUT_IMG_ELEMS + BATCH*TOPN)

__device__ __forceinline__ unsigned short f2bf(float f) {
  unsigned int u = __float_as_uint(f);
  u += 0x7fffu + ((u >> 16) & 1u);      // round-to-nearest-even
  return (unsigned short)(u >> 16);
}
__device__ __forceinline__ int imin(int a, int b) { return a < b ? a : b; }
__device__ __forceinline__ int imax(int a, int b) { return a > b ? a : b; }

// -------- pack activation: NCHW f32 -> NHWC bf16  Xt[b][s][c] --------------
__global__ __launch_bounds__(256) void k_pack_xt(const float* __restrict__ rpn,
                                                 unsigned short* __restrict__ Xt) {
  int idx = blockIdx.x * 256 + threadIdx.x;          // over 16*196*2048
  if (idx >= BATCH * HW1 * CIN1) return;
  int c   = idx & (CIN1 - 1);
  int pos = idx >> 11;                               // b*196 + s
  int b   = pos / HW1;
  int s   = pos - b * HW1;
  Xt[idx] = f2bf(rpn[((size_t)b * CIN1 + c) * HW1 + s]);
}

// -------- pack weights: OIHW f32 -> bf16 Wr[n][tap*2048 + c] ---------------
__global__ __launch_bounds__(256) void k_pack_wr(const float* __restrict__ w1,
                                                 unsigned short* __restrict__ Wr) {
  int idx = blockIdx.x * 256 + threadIdx.x;          // over 128*18432
  if (idx >= NCH * KDIM) return;
  int n   = idx / KDIM;
  int rem = idx - n * KDIM;
  int tap = rem >> 11;                               // 0..8
  int c   = rem & (CIN1 - 1);
  Wr[idx] = f2bf(w1[(size_t)n * KDIM + c * 9 + tap]);
}

// ---------------- conv1 as implicit-GEMM bf16 WMMA -------------------------
// Block: 256 threads (8 waves) = 64 M-rows x N=128. Wave w: M-tile (w>>1),
// N-tiles [(w&1)*4 .. +3]. 576 K-steps of 32, tap-major, double-buffered LDS.
#define ASTR 40   // LDS row stride (halves): 80B rows, 16B-aligned chunks
#define BSTR 40

__global__ __launch_bounds__(256) void k_conv1_wmma(
    const unsigned short* __restrict__ Xt,    // [16,196,2048] bf16 NHWC
    const unsigned short* __restrict__ Wr,    // [128][18432]  bf16 tap-major
    const float* __restrict__ bias,           // [128]
    float* __restrict__ d1)                   // [16,128,14,14] f32 (relu)
{
  __shared__ __attribute__((aligned(32))) unsigned short As[2][64 * ASTR];
  __shared__ __attribute__((aligned(32))) unsigned short Bs[2][128 * BSTR];

  const int tid  = threadIdx.x;
  const int wave = tid >> 5;
  const int lane = tid & 31;
  const int m0   = blockIdx.x * 64;
  const int mt   = wave >> 1;             // 0..3
  const int ntB  = (wave & 1) * 4;        // 0 or 4
  const int l16  = lane & 15;
  const int half = lane >> 4;

  v8f acc[4] = {};

  // A cooperative-load mapping (fixed per thread): row & 8-half chunk
  const int aLdRow = tid >> 2;            // 0..63
  const int aLdK0  = (tid & 3) * 8;       // 0,8,16,24
  const int mRow   = m0 + aLdRow;
  const int bIdx   = mRow / HW1;
  const int sIdx   = mRow - bIdx * HW1;
  const int oy     = sIdx / 14;
  const int ox     = sIdx - oy * 14;
  const unsigned short* XtB = Xt + (size_t)bIdx * HW1 * CIN1;
  // B cooperative-load mapping: 16 contiguous halves per thread
  const int bLdN   = tid >> 1;            // 0..127
  const int bLdK0  = (tid & 1) * 16;      // 0 or 16
  const unsigned short* WrN = Wr + (size_t)bLdN * KDIM;

  const int aRow = mt * 16 + l16;         // fragment row in LDS A

  u16x8 aR, bR0, bR1;
  auto load_tile = [&](int kt) {
    const int tap = kt >> 6;              // 0..8
    const int c0  = (kt & 63) * 32;
    const int dy  = tap / 3;
    const int dx  = tap - dy * 3;
    const int iy  = oy + dy - 1;
    const int ix  = ox + dx - 1;
    u16x8 z = {0, 0, 0, 0, 0, 0, 0, 0};
    aR = z;
    if ((unsigned)iy < 14u && (unsigned)ix < 14u)
      aR = *(const u16x8*)(XtB + (size_t)(iy * 14 + ix) * CIN1 + c0 + aLdK0);
    const unsigned short* bp = WrN + tap * CIN1 + c0 + bLdK0;
    bR0 = *(const u16x8*)bp;
    bR1 = *(const u16x8*)(bp + 8);
  };
  auto store_tile = [&](int p) {
    *(u16x8*)&As[p][aLdRow * ASTR + aLdK0]    = aR;
    *(u16x8*)&Bs[p][bLdN * BSTR + bLdK0]      = bR0;
    *(u16x8*)&Bs[p][bLdN * BSTR + bLdK0 + 8]  = bR1;
  };

  load_tile(0);
  store_tile(0);
  __syncthreads();

  for (int kt = 0; kt < KSTEPS; ++kt) {
    const int p = kt & 1;
    if (kt + 1 < KSTEPS) load_tile(kt + 1);   // overlap globals with compute

    // ---- A fragment: elems 0..7 = K half*8..+7 ; 8..15 = 16+half*8..+7 ----
    v16bf afrag;
    {
      const v8bf lo = *(const v8bf*)&As[p][aRow * ASTR + half * 8];
      const v8bf hi = *(const v8bf*)&As[p][aRow * ASTR + 16 + half * 8];
      #pragma unroll
      for (int i = 0; i < 8; ++i) { afrag[i] = lo[i]; afrag[i + 8] = hi[i]; }
    }
    // ---- 4 N-tiles: B frag K = half*8..+7, 16+half*8..+7 (matches A) -------
    #pragma unroll
    for (int j = 0; j < 4; ++j) {
      const int n = (ntB + j) * 16 + l16;
      v16bf bfrag;
      const v8bf blo = *(const v8bf*)&Bs[p][n * BSTR + half * 8];
      const v8bf bhi = *(const v8bf*)&Bs[p][n * BSTR + 16 + half * 8];
      #pragma unroll
      for (int i = 0; i < 8; ++i) { bfrag[i] = blo[i]; bfrag[i + 8] = bhi[i]; }
      acc[j] = __builtin_amdgcn_wmma_f32_16x16x32_bf16(
          false, afrag, false, bfrag, (short)0, acc[j], false, false);
    }

    if (kt + 1 < KSTEPS) store_tile(1 - p);   // fill the other buffer
    __syncthreads();
  }

  // ---- epilogue: bias + relu, scatter to NCHW ----
  #pragma unroll
  for (int j = 0; j < 4; ++j) {
    const int n  = (ntB + j) * 16 + l16;
    const float bn = bias[n];
    #pragma unroll
    for (int r = 0; r < 8; ++r) {
      int m = m0 + mt * 16 + half * 8 + r;
      int b = m / HW1;
      int s = m - b * HW1;
      float v = acc[j][r] + bn;
      d1[((size_t)b * NCH + n) * HW1 + s] = fmaxf(v, 0.0f);
    }
  }
}

// ---------------- small strided 3x3 convs (128->128, stride 2, pad 1) ------
__global__ __launch_bounds__(256) void k_conv_s2(
    const float* __restrict__ in, const float* __restrict__ w,
    const float* __restrict__ bias, float* __restrict__ out,
    int Hin, int Hout) {
  int idx = blockIdx.x * 256 + threadIdx.x;
  int total = BATCH * NCH * Hout * Hout;
  if (idx >= total) return;
  int ox = idx % Hout;
  int t  = idx / Hout;
  int oyv = t % Hout;
  t /= Hout;
  int co = t % NCH;
  int b  = t / NCH;
  float acc = bias[co];
  const float* wp = w + (size_t)co * NCH * 9;
  const float* ip = in + (size_t)b * NCH * Hin * Hin;
  for (int ci = 0; ci < NCH; ++ci) {
    #pragma unroll
    for (int dy = 0; dy < 3; ++dy) {
      int iy = oyv * 2 + dy - 1;
      if ((unsigned)iy >= (unsigned)Hin) continue;
      #pragma unroll
      for (int dx = 0; dx < 3; ++dx) {
        int ix = ox * 2 + dx - 1;
        if ((unsigned)ix >= (unsigned)Hin) continue;
        acc += ip[(ci * Hin + iy) * Hin + ix] * wp[ci * 9 + dy * 3 + dx];
      }
    }
  }
  out[idx] = fmaxf(acc, 0.0f);
}

// ---------------- tidy 1x1 convs -> concatenated scores --------------------
__global__ __launch_bounds__(256) void k_tidy(
    const float* __restrict__ d1, const float* __restrict__ d2,
    const float* __restrict__ d3,
    const float* __restrict__ wt1, const float* __restrict__ bt1,
    const float* __restrict__ wt2, const float* __restrict__ bt2,
    const float* __restrict__ wt3, const float* __restrict__ bt3,
    float* __restrict__ scores) {
  int idx = blockIdx.x * 256 + threadIdx.x;
  if (idx >= BATCH * NANCH) return;
  int b = idx / NANCH;
  int r = idx - b * NANCH;
  const float* feat; const float* wp; float bv; int stride;
  if (r < 1176) {
    int ch = r / 196, s = r - ch * 196;
    feat = d1 + (size_t)b * NCH * 196 + s; stride = 196;
    wp = wt1 + ch * NCH; bv = bt1[ch];
  } else if (r < 1470) {
    int q = r - 1176; int ch = q / 49, s = q - ch * 49;
    feat = d2 + (size_t)b * NCH * 49 + s; stride = 49;
    wp = wt2 + ch * NCH; bv = bt2[ch];
  } else {
    int q = r - 1470; int ch = q / 16, s = q - ch * 16;
    feat = d3 + (size_t)b * NCH * 16 + s; stride = 16;
    wp = wt3 + ch * NCH; bv = bt3[ch];
  }
  float acc = bv;
  for (int c = 0; c < NCH; ++c) acc += feat[c * stride] * wp[c];
  scores[idx] = acc;
}

// ---------------- anchors (fixed constants, padded, int) -------------------
__global__ __launch_bounds__(256) void k_anchors(int4* __restrict__ anchors) {
  int r = blockIdx.x * 256 + threadIdx.x;
  if (r >= NANCH) return;
  const float sc13 = 1.2599210498948732f;   // 2^(1/3)
  const float sc23 = 1.5874010519681994f;   // 2^(2/3)
  float strd, size, sc, ar; int iy, ix;
  const float ars[3] = {0.667f, 1.0f, 1.5f};
  if (r < 1176) {
    strd = 32.f; size = 48.f;
    int type = r / 196, cell = r - type * 196;
    sc = (type / 3 == 0) ? sc13 : sc23;
    ar = ars[type % 3];
    iy = cell / 14; ix = cell - iy * 14;
  } else if (r < 1470) {
    int q = r - 1176;
    strd = 64.f; size = 96.f;
    int type = q / 49, cell = q - type * 49;
    sc = (type / 3 == 0) ? sc13 : sc23;
    ar = ars[type % 3];
    iy = cell / 7; ix = cell - iy * 7;
  } else {
    int q = r - 1470;
    strd = 128.f; size = 192.f;
    int type = q / 16, cell = q - type * 16;
    int si = type / 3;
    sc = (si == 0) ? 1.0f : (si == 1 ? sc13 : sc23);
    ar = ars[type % 3];
    iy = cell / 4; ix = cell - iy * 4;
  }
  float cy = iy * strd + strd * 0.5f;
  float cx = ix * strd + strd * 0.5f;
  float h  = size * sc / sqrtf(ar);
  float w  = size * sc * sqrtf(ar);
  int4 a;
  a.x = (int)(cy - h * 0.5f + (float)PADV);
  a.y = (int)(cx - w * 0.5f + (float)PADV);
  a.z = (int)(cy + h * 0.5f + (float)PADV);
  a.w = (int)(cx + w * 0.5f + (float)PADV);
  anchors[r] = a;
}

// ---------------- greedy hard-NMS (top-6), IoU on the fly ------------------
__global__ __launch_bounds__(256) void k_nms(
    const float* __restrict__ scores, const int4* __restrict__ anchors,
    int* __restrict__ topidx, float* __restrict__ out_idx_f,
    float* __restrict__ out_prob) {
  __shared__ float s[NANCH];
  __shared__ float rv[256];
  __shared__ int   ri[256];
  __shared__ int   chosen;
  const int b = blockIdx.x, t = threadIdx.x;
  for (int j = t; j < NANCH; j += 256) s[j] = scores[b * NANCH + j];
  __syncthreads();
  for (int it = 0; it < TOPN; ++it) {
    float bv = -__builtin_inff(); int bi = 0x7fffffff;
    for (int j = t; j < NANCH; j += 256) {
      float v = s[j];
      if (v > bv || (v == bv && j < bi)) { bv = v; bi = j; }
    }
    rv[t] = bv; ri[t] = bi;
    __syncthreads();
    for (int off = 128; off > 0; off >>= 1) {
      if (t < off) {
        float v2 = rv[t + off]; int i2 = ri[t + off];
        if (v2 > rv[t] || (v2 == rv[t] && i2 < ri[t])) { rv[t] = v2; ri[t] = i2; }
      }
      __syncthreads();
    }
    if (t == 0) {
      chosen = ri[0];
      topidx[b * TOPN + it]    = ri[0];
      out_idx_f[b * TOPN + it] = (float)ri[0];
      out_prob[b * TOPN + it]  = rv[0];
    }
    __syncthreads();
    const int4 ca = anchors[chosen];
    const float ay0 = (float)ca.x, ax0 = (float)ca.y;
    const float ay1 = (float)ca.z, ax1 = (float)ca.w;
    const float areaA = (ay1 - ay0) * (ax1 - ax0);
    for (int j = t; j < NANCH; j += 256) {
      int4 cb = anchors[j];
      float by0 = (float)cb.x, bx0 = (float)cb.y;
      float by1 = (float)cb.z, bx1 = (float)cb.w;
      float l0 = fminf(ay1, by1) - fmaxf(ay0, by0);
      float l1 = fminf(ax1, bx1) - fmaxf(ax0, bx0);
      float inter = (l0 < 0.f || l1 < 0.f) ? 0.f : l0 * l1;
      float areaB = (by1 - by0) * (bx1 - bx0);
      float iou = inter / (areaA + areaB - inter);
      if (iou >= 0.25f) s[j] = -__builtin_inff();
    }
    __syncthreads();
  }
}

// ---------------- bilinear crop-resize with virtual padding ----------------
__device__ __forceinline__ float fetch_pad(const float* __restrict__ img,
                                           int b, int c, int yy, int xx) {
  int iy = yy - PADV, ix = xx - PADV;
  if ((unsigned)iy >= (unsigned)IMG || (unsigned)ix >= (unsigned)IMG) return 0.0f;
  return img[(((size_t)b * 3 + c) * IMG + iy) * IMG + ix];
}

__global__ __launch_bounds__(256) void k_crop(
    const float* __restrict__ image, const int4* __restrict__ anchors,
    const int* __restrict__ topidx, float* __restrict__ out) {
  const int iy = blockIdx.x;   // 0..223 output row
  const int r  = blockIdx.y;   // 0..5
  const int b  = blockIdx.z;   // 0..15
  const int4 box = anchors[topidx[b * TOPN + r]];
  const float by0 = (float)box.x, bx0 = (float)box.y;
  const float by1 = (float)box.z, bx1 = (float)box.w;
  const float ty = (float)iy / 223.0f;
  const float ys = by0 + ty * (by1 - by0 - 1.0f);
  const float yif = floorf(ys);
  const float fy = ys - yif;
  const int y0 = imin(imax((int)yif, 0), HP - 1);
  const int y1 = imin(y0 + 1, HP - 1);
  for (int j = threadIdx.x; j < 3 * CROP; j += 256) {
    int c  = j / CROP;
    int ix = j - c * CROP;
    float tx = (float)ix / 223.0f;
    float xs = bx0 + tx * (bx1 - bx0 - 1.0f);
    float xif = floorf(xs);
    float fx = xs - xif;
    int x0 = imin(imax((int)xif, 0), HP - 1);
    int x1 = imin(x0 + 1, HP - 1);
    float v00 = fetch_pad(image, b, c, y0, x0);
    float v01 = fetch_pad(image, b, c, y0, x1);
    float v10 = fetch_pad(image, b, c, y1, x0);
    float v11 = fetch_pad(image, b, c, y1, x1);
    float val = v00 * (1.f - fy) * (1.f - fx) + v01 * (1.f - fy) * fx
              + v10 * fy * (1.f - fx)         + v11 * fy * fx;
    out[((((size_t)b * TOPN + r) * 3 + c) * CROP + iy) * CROP + ix] = val;
  }
}

// ---------------------------------------------------------------------------
extern "C" void kernel_launch(void* const* d_in, const int* in_sizes, int n_in,
                              void* d_out, int out_size, void* d_ws, size_t ws_size,
                              hipStream_t stream) {
  const float* image = (const float*)d_in[0];
  const float* rpn   = (const float*)d_in[1];
  const float* w1  = (const float*)d_in[2];  const float* b1  = (const float*)d_in[3];
  const float* w2  = (const float*)d_in[4];  const float* b2  = (const float*)d_in[5];
  const float* w3  = (const float*)d_in[6];  const float* b3  = (const float*)d_in[7];
  const float* wt1 = (const float*)d_in[8];  const float* bt1 = (const float*)d_in[9];
  const float* wt2 = (const float*)d_in[10]; const float* bt2 = (const float*)d_in[11];
  const float* wt3 = (const float*)d_in[12]; const float* bt3 = (const float*)d_in[13];
  float* out = (float*)d_out;

  char* ws = (char*)d_ws;
  size_t off = 0;
  auto walloc = [&](size_t bytes) -> void* {
    void* p = ws + off;
    off = (off + bytes + 255) & ~(size_t)255;
    return p;
  };
  unsigned short* Wr = (unsigned short*)walloc((size_t)NCH * KDIM * 2);         // 4.7 MB
  unsigned short* Xt = (unsigned short*)walloc((size_t)BATCH * HW1 * CIN1 * 2); // 12.8 MB
  float* d1      = (float*)walloc((size_t)BATCH * NCH * HW1 * 4);
  float* d2      = (float*)walloc((size_t)BATCH * NCH * 49 * 4);
  float* d3      = (float*)walloc((size_t)BATCH * NCH * 16 * 4);
  float* scores  = (float*)walloc((size_t)BATCH * NANCH * 4);
  int4*  anchors = (int4*)walloc((size_t)NANCH * 16);
  int*   topidx  = (int*)walloc((size_t)BATCH * TOPN * 4);
  (void)ws_size; (void)in_sizes; (void)n_in; (void)out_size;

  const int nW = NCH * KDIM;            // 2359296
  const int nX = BATCH * HW1 * CIN1;    // 6422528
  k_pack_wr<<<(nW + 255) / 256, 256, 0, stream>>>(w1, Wr);
  k_pack_xt<<<(nX + 255) / 256, 256, 0, stream>>>(rpn, Xt);

  k_conv1_wmma<<<MROWS / 64, 256, 0, stream>>>(Xt, Wr, b1, d1);

  k_conv_s2<<<(BATCH * NCH * 49 + 255) / 256, 256, 0, stream>>>(d1, w2, b2, d2, 14, 7);
  k_conv_s2<<<(BATCH * NCH * 16 + 255) / 256, 256, 0, stream>>>(d2, w3, b3, d3, 7, 4);

  k_tidy<<<(BATCH * NANCH + 255) / 256, 256, 0, stream>>>(
      d1, d2, d3, wt1, bt1, wt2, bt2, wt3, bt3, scores);

  k_anchors<<<(NANCH + 255) / 256, 256, 0, stream>>>(anchors);

  k_nms<<<BATCH, 256, 0, stream>>>(scores, anchors, topidx,
                                   out + OUT_IDX_OFF, out + OUT_PROB_OFF);

  k_crop<<<dim3(CROP, TOPN, BATCH), 256, 0, stream>>>(image, anchors, topidx, out);
}
